// EncoderLayer_1924145349281
// MI455X (gfx1250) — matrix operand order repaired and checked by reading, not compiled, not run on hardware
//
#include <hip/hip_runtime.h>
#include <stdint.h>

// ---------------------------------------------------------------------------
// Types for CDNA5 WMMA (gfx1250, wave32)
// ---------------------------------------------------------------------------
typedef __attribute__((ext_vector_type(16))) __bf16 v16bf;
typedef __attribute__((ext_vector_type(8)))  float  v8f;

union Frag {
    v16bf    v;
    uint16_t h[16];
    uint4    q[2];
};

__device__ __forceinline__ uint16_t f2bf(float f) {
    union { float f; uint32_t u; } x;
    x.f = f;
    uint32_t u = x.u + 0x7FFFu + ((x.u >> 16) & 1u);  // round-to-nearest-even
    return (uint16_t)(u >> 16);
}

// ---------------------------------------------------------------------------
// Elementwise f32 -> bf16 conversion
// ---------------------------------------------------------------------------
__global__ __launch_bounds__(256) void conv_f32_bf16(const float* __restrict__ in,
                                                     uint16_t* __restrict__ out,
                                                     long long n) {
    long long i = (long long)blockIdx.x * blockDim.x + threadIdx.x;
    long long stride = (long long)gridDim.x * blockDim.x;
    for (; i < n; i += stride) out[i] = f2bf(in[i]);
}

// ---------------------------------------------------------------------------
// Generic bf16 WMMA GEMM:  C[m,n] = sum_k A[m,k] * B[n,k]   (B is "N x K")
// Block tile 128x64, BK=32, 128 threads (4 waves).
// Wave w computes rows [w*32, w*32+32) x all 64 cols:
//   2 A fragments + 4 B fragments -> 8 WMMAs per k-step (8 wmma : 12 ds_b128).
// Epilogues fused via template parameter.
// ---------------------------------------------------------------------------
#define EPI_QK     0   // out bf16 scatter [b,h,s,64]    (Q and K projections)
#define EPI_V      1   // out bf16 scatter [b,h,64,s]    (V projection, transposed)
#define EPI_SCORES 2   // outF = acc*0.125 + prev        (fp32, per-z batched)
#define EPI_WO     3   // outF = acc+bias ; outB = bf16(same)
#define EPI_FF1    4   // outB = bf16(gelu(acc+bias))
#define EPI_FF2    5   // outF = e0 + outF + acc + bias  (residuals, in-place)

template <int EPI>
__global__ __launch_bounds__(128) void gemm_bf16_wmma(
    const uint16_t* __restrict__ A,  long long sAz, int lda,
    const uint16_t* __restrict__ Bm, long long sBz, int ldb,
    int M, int N, int K,
    const float* __restrict__ bias,
    const float* __restrict__ e0,
    float*       __restrict__ outF,
    uint16_t*    __restrict__ outB)
{
    __shared__ uint16_t lA[128 * 40];  // stride 40 bf16 (80 B) -> 16B aligned, bank-skewed
    __shared__ uint16_t lB[64 * 40];

    const int tid  = threadIdx.x;
    const int lane = tid & 31;
    const int w    = tid >> 5;
    const int z    = blockIdx.z;
    const int m0   = blockIdx.y * 128;
    const int n0   = blockIdx.x * 64;

    A  += (long long)z * sAz;
    Bm += (long long)z * sBz;

    v8f c[2][4];
    #pragma unroll
    for (int u = 0; u < 2; ++u)
        #pragma unroll
        for (int t = 0; t < 4; ++t)
            c[u][t] = (v8f){0.f, 0.f, 0.f, 0.f, 0.f, 0.f, 0.f, 0.f};

    const int mm = lane & 15;   // row/col within 16-tile
    const int kh = lane >> 4;   // K-half select (ISA 16-bit operand layout)

    for (int k0 = 0; k0 < K; k0 += 32) {
        // cooperative load: A 128x32 (512 chunks) + B 64x32 (256 chunks), 16B each
        #pragma unroll
        for (int i = 0; i < 4; ++i) {
            int id = tid + i * 128;        // 0..511
            int r  = id >> 2;              // 0..127
            int cc = (id & 3) << 3;        // 0,8,16,24
            *reinterpret_cast<uint4*>(&lA[r * 40 + cc]) =
                *reinterpret_cast<const uint4*>(A + (long long)(m0 + r) * lda + k0 + cc);
        }
        #pragma unroll
        for (int i = 0; i < 2; ++i) {
            int id = tid + i * 128;        // 0..255
            int r  = id >> 2;              // 0..63
            int cc = (id & 3) << 3;
            *reinterpret_cast<uint4*>(&lB[r * 40 + cc]) =
                *reinterpret_cast<const uint4*>(Bm + (long long)(n0 + r) * ldb + k0 + cc);
        }
        if (k0 + 32 < K) {                 // pull next tiles toward cache (global_prefetch_b8)
            __builtin_prefetch(A + (long long)(m0 + (tid >> 1)) * lda + k0 + 32 + ((tid & 1) << 4), 0, 0);
            __builtin_prefetch(Bm + (long long)(n0 + (tid >> 1)) * ldb + k0 + 32 + ((tid & 1) << 4), 0, 0);
        }
        __syncthreads();

        // A fragments: lane holds M=mm, K = kh*8+{0..7} then 16+kh*8+{0..7}
        Frag a[2];
        #pragma unroll
        for (int u = 0; u < 2; ++u) {
            int ar = (w * 32 + u * 16 + mm) * 40;
            a[u].q[0] = *reinterpret_cast<const uint4*>(&lA[ar + kh * 8]);
            a[u].q[1] = *reinterpret_cast<const uint4*>(&lA[ar + 16 + kh * 8]);
        }

        #pragma unroll
        for (int t = 0; t < 4; ++t) {
            // B fragment: lane holds N=mm(+t*16), K = kh*8+{0..7} then 16+kh*8+{0..7}
            Frag b;
            int nn = t * 16 + mm;
            int kb = kh * 8;
            b.q[0] = *reinterpret_cast<const uint4*>(&lB[nn * 40 + kb]);
            b.q[1] = *reinterpret_cast<const uint4*>(&lB[nn * 40 + kb + 16]);
            c[0][t] = __builtin_amdgcn_wmma_f32_16x16x32_bf16(false, a[0].v, false, b.v,
                                                              (short)0, c[0][t], false, false);
            c[1][t] = __builtin_amdgcn_wmma_f32_16x16x32_bf16(false, a[1].v, false, b.v,
                                                              (short)0, c[1][t], false, false);
        }
        __syncthreads();
    }

    // Epilogue: VGPR r of acc [u][t], lane L -> M = u*16 + r + 8*(L>>4), N = t*16 + (L&15)
    #pragma unroll
    for (int u = 0; u < 2; ++u) {
        #pragma unroll
        for (int t = 0; t < 4; ++t) {
            #pragma unroll
            for (int r = 0; r < 8; ++r) {
                int row = m0 + w * 32 + u * 16 + r + (kh << 3);
                int col = n0 + t * 16 + mm;
                float acc = c[u][t][r];
                if (EPI == EPI_QK || EPI == EPI_V) {
                    float val = acc + bias[col];
                    int b  = row >> 10;      // row / S
                    int s  = row & 1023;
                    int hh = col >> 6;       // col / 64
                    int d  = col & 63;
                    long long idx;
                    if (EPI == EPI_QK)
                        idx = ((long long)(b * 16 + hh) * 1024 + s) * 64 + d;  // [b,h,s,dk]
                    else
                        idx = ((long long)(b * 16 + hh) * 64 + d) * 1024 + s;  // [b,h,dk,s]
                    outB[idx] = f2bf(val);
                } else if (EPI == EPI_SCORES) {
                    long long o = (long long)z * M * N + (long long)row * N + col;
                    outF[o] = acc * 0.125f + e0[o];          // scale = 64^-0.5
                } else if (EPI == EPI_WO) {
                    float val = acc + bias[col];
                    long long o = (long long)row * N + col;
                    outF[o] = val;                            // src2 (fp32, lives in d_out)
                    outB[o] = f2bf(val);                      // src2 (bf16, FFN input)
                } else if (EPI == EPI_FF1) {
                    float val = acc + bias[col];
                    float g = 0.5f * val * (1.0f + erff(val * 0.70710678118f)); // exact GELU
                    long long o = (long long)row * N + col;
                    outB[o] = f2bf(g);
                } else { // EPI_FF2
                    float val = acc + bias[col];
                    long long o = (long long)row * N + col;
                    outF[o] = e0[o] + outF[o] + val;          // src + src2 + ff
                }
            }
        }
    }
}

// ---------------------------------------------------------------------------
// Fused softmax + P@V.  One block = one (b*H+h, 16-row i-tile).
// Phase 1: row max / sum(exp) in LDS. Phase 2: each of 4 waves builds the bf16
// P fragment on the fly (v_exp) and WMMA-accumulates against V^T from global.
// ---------------------------------------------------------------------------
__global__ __launch_bounds__(128) void attn_softmax_av(
    const float*    __restrict__ scores,   // [B*H, S, S] fp32 (in d_out)
    const uint16_t* __restrict__ vt,       // [B*H, 64, S] bf16
    uint16_t*       __restrict__ ctx)      // [B, S, H*64] bf16
{
    __shared__ float red[16 * 8];
    __shared__ float rmax[16];
    __shared__ float rinv[16];

    const int tid  = threadIdx.x;
    const int lane = tid & 31;
    const int w    = tid >> 5;
    const int z    = blockIdx.y;          // b*H + h
    const int i0   = blockIdx.x * 16;

    const float* srow = scores + ((long long)z * 1024 + i0) * 1024;

    // ---- phase 1: per-row max and 1/sum(exp) ----
    {
        int row  = tid >> 3;              // 0..15
        int part = tid & 7;               // 0..7, 128 cols each
        const float* sr = srow + (long long)row * 1024 + part * 128;
        float mx = -3.4e38f;
        for (int cc = 0; cc < 128; ++cc) mx = fmaxf(mx, sr[cc]);
        red[row * 8 + part] = mx;
        __syncthreads();
        if (part == 0) {
            float m = red[row * 8];
            for (int p = 1; p < 8; ++p) m = fmaxf(m, red[row * 8 + p]);
            rmax[row] = m;
        }
        __syncthreads();
        float m = rmax[row];
        float sum = 0.0f;
        for (int cc = 0; cc < 128; ++cc) sum += __expf(sr[cc] - m);
        red[row * 8 + part] = sum;
        __syncthreads();
        if (part == 0) {
            float s = 0.0f;
            for (int p = 0; p < 8; ++p) s += red[row * 8 + p];
            rinv[row] = 1.0f / s;
        }
        __syncthreads();
    }

    // ---- phase 2: wave w produces d-columns [w*16, w*16+16) ----
    const int mm = lane & 15;
    const int kh = lane >> 4;
    const float* srm = srow + (long long)mm * 1024;
    const float mval = rmax[mm];
    const float ivs  = rinv[mm];
    const uint16_t* vrow = vt + (long long)z * 64 * 1024 + (long long)(w * 16 + mm) * 1024;

    v8f c = (v8f){0.f, 0.f, 0.f, 0.f, 0.f, 0.f, 0.f, 0.f};
    for (int j0 = 0; j0 < 1024; j0 += 32) {
        Frag a;   // P fragment, same lane layout as GEMM A operand
        #pragma unroll
        for (int jj = 0; jj < 8; ++jj) {
            a.h[jj]     = f2bf(__expf(srm[j0 + kh * 8 + jj]      - mval) * ivs);
            a.h[8 + jj] = f2bf(__expf(srm[j0 + 16 + kh * 8 + jj] - mval) * ivs);
        }
        Frag b;   // V^T fragment: lane holds N=d, K = j0 + kh*8+{0..7} then j0+16+kh*8+{0..7}
        int kb = kh * 8;
        b.q[0] = *reinterpret_cast<const uint4*>(vrow + j0 + kb);
        b.q[1] = *reinterpret_cast<const uint4*>(vrow + j0 + kb + 16);
        c = __builtin_amdgcn_wmma_f32_16x16x32_bf16(false, a.v, false, b.v,
                                                    (short)0, c, false, false);
    }

    const int b  = z >> 4;
    const int hh = z & 15;
    #pragma unroll
    for (int r = 0; r < 8; ++r) {
        int si  = i0 + r + (kh << 3);
        int col = hh * 64 + w * 16 + mm;
        ctx[((long long)b * 1024 + si) * 1024 + col] = f2bf(c[r]);
    }
}

// ---------------------------------------------------------------------------
// Host-side orchestration
// ---------------------------------------------------------------------------
extern "C" void kernel_launch(void* const* d_in, const int* in_sizes, int n_in,
                              void* d_out, int out_size, void* d_ws, size_t ws_size,
                              hipStream_t stream)
{
    const float* src  = (const float*)d_in[0];
    const float* prev = (const float*)d_in[1];
    const float* Wq   = (const float*)d_in[2];
    const float* bq   = (const float*)d_in[3];
    const float* Wk   = (const float*)d_in[4];
    const float* bk   = (const float*)d_in[5];
    const float* Wv   = (const float*)d_in[6];
    const float* bv   = (const float*)d_in[7];
    const float* Wo   = (const float*)d_in[8];
    const float* bo   = (const float*)d_in[9];
    const float* W1   = (const float*)d_in[10];
    const float* b1   = (const float*)d_in[11];
    const float* W2   = (const float*)d_in[12];
    const float* b2   = (const float*)d_in[13];

    float* outF   = (float*)d_out;                               // out_final [8192,1024]
    float* scores = outF + (long long)8 * 1024 * 1024;           // scores [128,1024,1024]

    uint8_t* ws = (uint8_t*)d_ws;                                // all offsets 16B aligned
    uint16_t* srcb = (uint16_t*)(ws);                            // 16 MiB
    uint16_t* Wqb  = (uint16_t*)(ws + 16777216);                 // 2 MiB
    uint16_t* Wkb  = (uint16_t*)(ws + 18874368);
    uint16_t* Wvb  = (uint16_t*)(ws + 20971520);
    uint16_t* Wob  = (uint16_t*)(ws + 23068672);
    uint16_t* W1b  = (uint16_t*)(ws + 25165824);                 // 8 MiB
    uint16_t* W2b  = (uint16_t*)(ws + 33554432);                 // 8 MiB
    uint16_t* qb   = (uint16_t*)(ws + 41943040);                 // 16 MiB [b,h,s,64]
    uint16_t* kb_  = (uint16_t*)(ws + 58720256);                 // 16 MiB [b,h,s,64]
    uint16_t* vtb  = (uint16_t*)(ws + 75497472);                 // 16 MiB [b,h,64,s]
    uint16_t* ctxb = (uint16_t*)(ws + 92274688);                 // 16 MiB [b,s,1024]
    uint16_t* s2b  = (uint16_t*)(ws + 109051904);                // 16 MiB src2 bf16
    uint16_t* ff1b = (uint16_t*)(ws + 125829120);                // 64 MiB [8192,4096]

    // f32 -> bf16 staging
    conv_f32_bf16<<<2048, 256, 0, stream>>>(src, srcb, 8LL * 1024 * 1024);
    conv_f32_bf16<<<1024, 256, 0, stream>>>(Wq, Wqb, 1024LL * 1024);
    conv_f32_bf16<<<1024, 256, 0, stream>>>(Wk, Wkb, 1024LL * 1024);
    conv_f32_bf16<<<1024, 256, 0, stream>>>(Wv, Wvb, 1024LL * 1024);
    conv_f32_bf16<<<1024, 256, 0, stream>>>(Wo, Wob, 1024LL * 1024);
    conv_f32_bf16<<<2048, 256, 0, stream>>>(W1, W1b, 4096LL * 1024);
    conv_f32_bf16<<<2048, 256, 0, stream>>>(W2, W2b, 4096LL * 1024);

    dim3 blk(128);

    // Q/K/V projections: [8192,1024] x [1024,1024]^T   (block tile 128x64)
    gemm_bf16_wmma<EPI_QK><<<dim3(16, 64, 1), blk, 0, stream>>>(
        srcb, 0, 1024, Wqb, 0, 1024, 8192, 1024, 1024, bq, nullptr, nullptr, qb);
    gemm_bf16_wmma<EPI_QK><<<dim3(16, 64, 1), blk, 0, stream>>>(
        srcb, 0, 1024, Wkb, 0, 1024, 8192, 1024, 1024, bk, nullptr, nullptr, kb_);
    gemm_bf16_wmma<EPI_V><<<dim3(16, 64, 1), blk, 0, stream>>>(
        srcb, 0, 1024, Wvb, 0, 1024, 8192, 1024, 1024, bv, nullptr, nullptr, vtb);

    // scores = Q K^T * scale + prev, per (b,h) batch in grid.z
    gemm_bf16_wmma<EPI_SCORES><<<dim3(16, 8, 128), blk, 0, stream>>>(
        qb, 1024LL * 64, 64, kb_, 1024LL * 64, 64, 1024, 1024, 64,
        nullptr, prev, scores, nullptr);

    // softmax + P@V -> ctx [b,s,1024]
    attn_softmax_av<<<dim3(64, 128, 1), blk, 0, stream>>>(scores, vtb, ctxb);

    // src2 = ctx @ Wo^T + bo  (fp32 into d_out final region, bf16 copy for FFN)
    gemm_bf16_wmma<EPI_WO><<<dim3(16, 64, 1), blk, 0, stream>>>(
        ctxb, 0, 1024, Wob, 0, 1024, 8192, 1024, 1024, bo, nullptr, outF, s2b);

    // ff1 = gelu(src2 @ W1^T + b1)
    gemm_bf16_wmma<EPI_FF1><<<dim3(64, 64, 1), blk, 0, stream>>>(
        s2b, 0, 1024, W1b, 0, 1024, 8192, 4096, 1024, b1, nullptr, nullptr, ff1b);

    // out_final = src + src2 + (ff1 @ W2^T + b2)   (src2 read from d_out, in place)
    gemm_bf16_wmma<EPI_FF2><<<dim3(16, 64, 1), blk, 0, stream>>>(
        ff1b, 0, 4096, W2b, 0, 4096, 8192, 1024, 4096, b2, src, outF, nullptr);
}
